// SVDiffModule_49589692400414
// MI455X (gfx1250) — matrix operand order repaired and checked by reading, not compiled, not run on hardware
//
#include <hip/hip_runtime.h>

// out[M,N] = (U * relu(S + delta)) @ Vh
// M=4096, R(K)=4096, N=16384, fp32 in/out.
// Compute-bound (AI ~955 FLOP/B): bf16x3 split-precision WMMA core
// (hi*hi + hi*lo + lo*hi, fp32 accumulate) at ~3/8 the instruction cost
// of the native f32 16x16x4 WMMA path, ~2^-16 relative error.
// Double-buffered LDS (120KB of the WGP's 320KB): one barrier per K-step,
// global prefetch hides under the WMMA phase.

#define MDIM 4096
#define KDIM 4096
#define NDIM 16384

#define BM 128
#define BN 256
#define BK 32
#define LDT 40  // padded LDS row stride in bf16 elements (80B: 16B-aligned, bank-spread)
#define NTHREADS 512

#define A_SZ (BM * LDT)
#define B_SZ (BN * LDT)

typedef __attribute__((ext_vector_type(16))) __bf16       v16bf;
typedef __attribute__((ext_vector_type(8)))  float        v8f;
typedef __attribute__((ext_vector_type(4)))  float        f32x4;
typedef __attribute__((ext_vector_type(4)))  unsigned int u32x4;
typedef __attribute__((ext_vector_type(2)))  unsigned int u32x2;

struct PairU { unsigned int h, l; };

// One v_perm_b32: D[15:0] = lo_elem[31:16], D[31:16] = hi_elem[31:16]
__device__ __forceinline__ unsigned int packhi16(unsigned int hi_elem, unsigned int lo_elem) {
  return __builtin_amdgcn_perm(hi_elem, lo_elem, 0x07060302u);
}

// Truncation-based fp32 -> (bf16 hi, bf16 lo) split for a pair of elements.
// hi = x & 0xffff0000 (top 8 mantissa bits), lo = bf16_trunc(x - hi) (next 8 bits).
// No v_cvt needed: 2x v_and + paired sub (v_pk_add_f32) + 2x v_perm per 2 elements.
__device__ __forceinline__ PairU split2(float x0, float x1) {
  unsigned int b0 = __float_as_uint(x0);
  unsigned int b1 = __float_as_uint(x1);
  float h0 = __uint_as_float(b0 & 0xffff0000u);
  float h1 = __uint_as_float(b1 & 0xffff0000u);
  float l0 = x0 - h0;
  float l1 = x1 - h1;
  PairU r;
  r.h = packhi16(b1, b0);
  r.l = packhi16(__float_as_uint(l1), __float_as_uint(l0));
  return r;
}

// A-fragment (16x32 bf16, M x K): lane-group 0 holds K 0-7 then 16-23; group 1 holds 8-15 then 24-31.
__device__ __forceinline__ v16bf ldfragA(const __bf16* t, int row, int grp) {
  union { v16bf v; u32x4 q[2]; } f;
  const __bf16* p = t + row * LDT;
  f.q[0] = *(const u32x4*)(p + grp * 8);
  f.q[1] = *(const u32x4*)(p + 16 + grp * 8);
  return f.v;
}

// B-fragment (32x16 bf16, K x N), LDS holds B transposed [N][K]:
// lane-group 0 holds K 0-7 then 16-23; group 1 holds 8-15 then 24-31.
__device__ __forceinline__ v16bf ldfragB(const __bf16* t, int row, int grp) {
  union { v16bf v; u32x4 q[2]; } f;
  const __bf16* p = t + row * LDT + grp * 8;
  f.q[0] = *(const u32x4*)(p);
  f.q[1] = *(const u32x4*)(p + 16);
  return f.v;
}

__global__ __launch_bounds__(NTHREADS) void svdiff_wmma_gemm(
    const float* __restrict__ U, const float* __restrict__ S,
    const float* __restrict__ Vh, const float* __restrict__ delta,
    float* __restrict__ out)
{
  // double-buffered stages: [0] and [1]
  __shared__ __align__(16) __bf16 Ah[2 * A_SZ];
  __shared__ __align__(16) __bf16 Al[2 * A_SZ];
  __shared__ __align__(16) __bf16 Bh[2 * B_SZ];
  __shared__ __align__(16) __bf16 Bl[2 * B_SZ];

  const int tid  = threadIdx.x;
  const int lane = tid & 31;
  const int wave = tid >> 5;          // 0..15 (16 wave32s = 4 waves/SIMD in one WGP)
  const int wm   = (wave >> 3) * 64;  // 2 waves across M
  const int wn   = (wave & 7) * 32;   // 8 waves across N
  const int lr   = lane & 15;
  const int grp  = lane >> 4;

  const int bm = blockIdx.y * BM;
  const int bn = blockIdx.x * BN;

  // A staging: 8 threads x float4 across K=32, 64 threads across M, 2 M-steps of 64
  const int atk = (tid & 7) * 4;
  const int atm = tid >> 3;           // 0..63
  // B staging: 256 threads across N, 2 thread-groups x 16 across K (coalesced per K-row)
  const int btn = tid & 255;
  const int btk = (tid >> 8) * 16;

  f32x4 aReg[2];
  f32x4 sReg;
  float bReg[16];

  v8f zero = {};
  v8f acc[4][2];
  for (int i = 0; i < 4; ++i)
    for (int j = 0; j < 2; ++j)
      acc[i][j] = zero;

  // ---- prologue: load k-slab 0, convert, store into stage 0 ----
  {
    f32x4 sS = *(const f32x4*)(S + atk);
    f32x4 sD = *(const f32x4*)(delta + atk);
    for (int j = 0; j < 4; ++j) sReg[j] = fmaxf(sS[j] + sD[j], 0.0f);
    for (int i = 0; i < 2; ++i)
      aReg[i] = *(const f32x4*)(U + (size_t)(bm + atm + i * 64) * KDIM + atk);
    for (int j = 0; j < 16; ++j)
      bReg[j] = Vh[(size_t)(btk + j) * NDIM + bn + btn];

    for (int i = 0; i < 2; ++i) {
      float x0 = aReg[i][0] * sReg[0];
      float x1 = aReg[i][1] * sReg[1];
      float x2 = aReg[i][2] * sReg[2];
      float x3 = aReg[i][3] * sReg[3];
      PairU p01 = split2(x0, x1);
      PairU p23 = split2(x2, x3);
      u32x2 ph, pl;
      ph[0] = p01.h; ph[1] = p23.h;
      pl[0] = p01.l; pl[1] = p23.l;
      int row = atm + i * 64;
      *(u32x2*)(Ah + row * LDT + atk) = ph;
      *(u32x2*)(Al + row * LDT + atk) = pl;
    }
    for (int c = 0; c < 2; ++c) {
      u32x4 ph, pl;
      for (int q = 0; q < 4; ++q) {
        PairU pq = split2(bReg[c * 8 + q * 2], bReg[c * 8 + q * 2 + 1]);
        ph[q] = pq.h;
        pl[q] = pq.l;
      }
      *(u32x4*)(Bh + btn * LDT + btk + c * 8) = ph;
      *(u32x4*)(Bl + btn * LDT + btk + c * 8) = pl;
    }
  }
  __syncthreads();

  const int KSTEPS = KDIM / BK;
  for (int ks = 0; ks < KSTEPS; ++ks) {
    const int buf = ks & 1;
    const int nxt = buf ^ 1;
    const bool more = (ks + 1) < KSTEPS;

    // ---- issue global loads for the next K-slab (latency hides under WMMA) ----
    if (more) {
      const int k0 = (ks + 1) * BK;
      f32x4 sS = *(const f32x4*)(S + k0 + atk);
      f32x4 sD = *(const f32x4*)(delta + k0 + atk);
      for (int j = 0; j < 4; ++j) sReg[j] = fmaxf(sS[j] + sD[j], 0.0f);
      for (int i = 0; i < 2; ++i)
        aReg[i] = *(const f32x4*)(U + (size_t)(bm + atm + i * 64) * KDIM + k0 + atk);
      for (int j = 0; j < 16; ++j)
        bReg[j] = Vh[(size_t)(k0 + btk + j) * NDIM + bn + btn];
    }

    // ---- WMMA phase from stage `buf`: 4 M-tiles x 2 N-tiles x 3 terms = 24 wmma/wave ----
    {
      const __bf16* ahp = Ah + buf * A_SZ;
      const __bf16* alp = Al + buf * A_SZ;
      const __bf16* bhp = Bh + buf * B_SZ;
      const __bf16* blp = Bl + buf * B_SZ;
      v16bf ah[4], al[4];
      for (int mt = 0; mt < 4; ++mt) {
        int row = wm + mt * 16 + lr;
        ah[mt] = ldfragA(ahp, row, grp);
        al[mt] = ldfragA(alp, row, grp);
      }
      for (int nt = 0; nt < 2; ++nt) {
        int nrow = wn + nt * 16 + lr;
        v16bf bh = ldfragB(bhp, nrow, grp);
        v16bf bl = ldfragB(blp, nrow, grp);
        for (int mt = 0; mt < 4; ++mt) {
          acc[mt][nt] = __builtin_amdgcn_wmma_f32_16x16x32_bf16(
              false, ah[mt], false, bh, (short)0, acc[mt][nt], false, false);
          acc[mt][nt] = __builtin_amdgcn_wmma_f32_16x16x32_bf16(
              false, ah[mt], false, bl, (short)0, acc[mt][nt], false, false);
          acc[mt][nt] = __builtin_amdgcn_wmma_f32_16x16x32_bf16(
              false, al[mt], false, bh, (short)0, acc[mt][nt], false, false);
        }
      }
    }

    // ---- convert + store next slab into stage `nxt` (reads of `nxt` done at last barrier) ----
    if (more) {
      __bf16* ahp = Ah + nxt * A_SZ;
      __bf16* alp = Al + nxt * A_SZ;
      __bf16* bhp = Bh + nxt * B_SZ;
      __bf16* blp = Bl + nxt * B_SZ;
      for (int i = 0; i < 2; ++i) {
        float x0 = aReg[i][0] * sReg[0];
        float x1 = aReg[i][1] * sReg[1];
        float x2 = aReg[i][2] * sReg[2];
        float x3 = aReg[i][3] * sReg[3];
        PairU p01 = split2(x0, x1);
        PairU p23 = split2(x2, x3);
        u32x2 ph, pl;
        ph[0] = p01.h; ph[1] = p23.h;
        pl[0] = p01.l; pl[1] = p23.l;
        int row = atm + i * 64;
        *(u32x2*)(ahp + row * LDT + atk) = ph;
        *(u32x2*)(alp + row * LDT + atk) = pl;
      }
      for (int c = 0; c < 2; ++c) {
        u32x4 ph, pl;
        for (int q = 0; q < 4; ++q) {
          PairU pq = split2(bReg[c * 8 + q * 2], bReg[c * 8 + q * 2 + 1]);
          ph[q] = pq.h;
          pl[q] = pq.l;
        }
        *(u32x4*)(bhp + btn * LDT + btk + c * 8) = ph;
        *(u32x4*)(blp + btn * LDT + btk + c * 8) = pl;
      }
      __syncthreads();  // single barrier per K-step
    }
  }

  // ---- epilogue: C layout VGPR r -> M = r (lanes 0-15) / 8+r (lanes 16-31), N = lane&15.
  // Streaming output: non-temporal stores keep L2 for the reused U/Vh tiles.
  for (int mt = 0; mt < 4; ++mt) {
    for (int nt = 0; nt < 2; ++nt) {
      int n = bn + wn + nt * 16 + lr;
      int mbase = bm + wm + mt * 16 + grp * 8;
      float* p = out + (size_t)mbase * NDIM + n;
      for (int r = 0; r < 8; ++r) {
        __builtin_nontemporal_store(acc[mt][nt][r], p);
        p += NDIM;
      }
    }
  }
}

extern "C" void kernel_launch(void* const* d_in, const int* in_sizes, int n_in,
                              void* d_out, int out_size, void* d_ws, size_t ws_size,
                              hipStream_t stream) {
  (void)in_sizes; (void)n_in; (void)out_size; (void)d_ws; (void)ws_size;
  const float* U     = (const float*)d_in[0];
  const float* S     = (const float*)d_in[1];
  const float* Vh    = (const float*)d_in[2];
  const float* delta = (const float*)d_in[3];
  // d_in[4] (weight) unused: weight_type is None in the reference.
  float* out = (float*)d_out;

  dim3 grid(NDIM / BN, MDIM / BM);  // 64 x 32 workgroups
  dim3 block(NTHREADS);             // 16 wave32s -> 4 waves/SIMD on one WGP
  hipLaunchKernelGGL(svdiff_wmma_gemm, grid, block, 0, stream, U, S, Vh, delta, out);
}
